// TernaryFusedMoEMethod_2860448219904
// MI455X (gfx1250) — matrix operand order repaired and checked by reading, not compiled, not run on hardware
//
#include <hip/hip_runtime.h>

// MI455X / gfx1250 ternary fused MoE, wave32 + WMMA f16 path, round 2.
//   E=8 experts, T=4096 tokens, K=2048 hidden, I=768 intermediate, top-2.
// Strategy: token gather per expert (true top-2 sparsity, 4x compute saving),
// one-shot dequant of ternary weights to f16 (exact up to f16 rounding),
// GEMM inner loops are pure b128 loads + v_wmma_f32_16x16x32_f16.

typedef __attribute__((ext_vector_type(16))) _Float16 v16h;
typedef __attribute__((ext_vector_type(8)))  _Float16 v8h;
typedef __attribute__((ext_vector_type(4)))  _Float16 v4h;
typedef __attribute__((ext_vector_type(8)))  float    v8f;

#define E_ 8
#define T_ 4096
#define K_ 2048
#define I_ 768

#define WMMA_F16(A, B, C) \
  __builtin_amdgcn_wmma_f32_16x16x32_f16(false, (A), false, (B), (short)0, (C), false, false)

// ---------------------------------------------------------------------------
// Zero output + per-expert counters (ws/out are poisoned; init ourselves)
// ---------------------------------------------------------------------------
__global__ void zero_kernel(float* __restrict__ out, int n, int* __restrict__ counts) {
  int i = blockIdx.x * blockDim.x + threadIdx.x;
  if (i < n) out[i] = 0.f;
  if (i < E_) counts[i] = 0;
}

// ---------------------------------------------------------------------------
// Routing: softmax + top-2 + renormalize, build gathered per-expert lists.
// ---------------------------------------------------------------------------
__global__ void routing_kernel(const float* __restrict__ logits,
                               int* __restrict__ counts,
                               int* __restrict__ idx,
                               float* __restrict__ gateg) {
  int t = blockIdx.x * blockDim.x + threadIdx.x;
  if (t >= T_) return;
  float p[E_];
  float m = -3.0e38f;
  for (int e = 0; e < E_; ++e) { p[e] = logits[t * E_ + e]; m = fmaxf(m, p[e]); }
  for (int e = 0; e < E_; ++e) p[e] = __expf(p[e] - m);   // softmax denom cancels in renorm
  int i1 = 0; float p1 = p[0];
  for (int e = 1; e < E_; ++e) { if (p[e] > p1) { p1 = p[e]; i1 = e; } }
  int i2 = (i1 == 0) ? 1 : 0; float p2 = p[i2];
  for (int e = 0; e < E_; ++e) {
    if (e == i1) continue;
    if (p[e] > p2) { p2 = p[e]; i2 = e; }
  }
  float inv = __fdividef(1.f, p1 + p2);
  int s1 = atomicAdd(&counts[i1], 1);
  idx[i1 * T_ + s1] = t;  gateg[i1 * T_ + s1] = p1 * inv;
  int s2 = atomicAdd(&counts[i2], 1);
  idx[i2 * T_ + s2] = t;  gateg[i2 * T_ + s2] = p2 * inv;
}

// ---------------------------------------------------------------------------
// One-shot dequant: f16 W = (code - 1) * alpha[input_col].  4 elems/thread.
// Hoists ALL dequant VALU work out of the GEMM inner loops.
// ---------------------------------------------------------------------------
__global__ void dequant13_kernel(const int* __restrict__ w13,
                                 const float* __restrict__ alpha13,
                                 _Float16* __restrict__ w13h) {
  long j = ((long)blockIdx.x * blockDim.x + threadIdx.x) * 4;
  if (j >= (long)E_ * 2 * I_ * K_) return;
  int4 c = *(const int4*)(w13 + j);
  int k = (int)(j % K_);
  int e = (int)(j / ((long)2 * I_ * K_));
  float4 al = *(const float4*)(alpha13 + (size_t)e * K_ + k);
  v4h h = { (_Float16)((float)(c.x - 1) * al.x), (_Float16)((float)(c.y - 1) * al.y),
            (_Float16)((float)(c.z - 1) * al.z), (_Float16)((float)(c.w - 1) * al.w) };
  *(v4h*)(w13h + j) = h;
}

__global__ void dequant2_kernel(const int* __restrict__ w2,
                                const float* __restrict__ alpha2,
                                _Float16* __restrict__ w2h) {
  long j = ((long)blockIdx.x * blockDim.x + threadIdx.x) * 4;
  if (j >= (long)E_ * K_ * I_) return;
  int4 c = *(const int4*)(w2 + j);
  int i = (int)(j % I_);
  int e = (int)(j / ((long)K_ * I_));
  float4 al = *(const float4*)(alpha2 + (size_t)e * I_ + i);
  v4h h = { (_Float16)((float)(c.x - 1) * al.x), (_Float16)((float)(c.y - 1) * al.y),
            (_Float16)((float)(c.z - 1) * al.z), (_Float16)((float)(c.w - 1) * al.w) };
  *(v4h*)(w2h + j) = h;
}

__global__ void xcast_kernel(const float* __restrict__ x, _Float16* __restrict__ xh) {
  long j = ((long)blockIdx.x * blockDim.x + threadIdx.x) * 4;
  if (j >= (long)T_ * K_) return;
  float4 v = *(const float4*)(x + j);
  v4h h = { (_Float16)v.x, (_Float16)v.y, (_Float16)v.z, (_Float16)v.w };
  *(v4h*)(xh + j) = h;
}

// A-fragment assembly: code-chosen K order (not AMD's table order) for 16-bit A, lane group (lane/16)
// holds K = {half*16 .. half*16+15} (contiguous, matches B): two b128 loads.
__device__ __forceinline__ v16h load_a_frag(const _Float16* row, int kb, int half) {
  v8h lo = *(const v8h*)(row + kb + half * 16);
  v8h hi = *(const v8h*)(row + kb + half * 16 + 8);
  return __builtin_shufflevector(lo, hi, 0, 1, 2, 3, 4, 5, 6, 7,
                                 8, 9, 10, 11, 12, 13, 14, 15);
}

// ---------------------------------------------------------------------------
// GEMM1 (gathered): h = x @ W13_e^T, fused SiLU(g)*u -> act (f16).
// Block = 64 (2 waves); each wave: 32 slots x 16 gate/up column pair
// (2 token tiles share each B fragment -> 4 WMMA per 8 b128 loads).
// Grid: (T/32, I/32, E); wave-uniform early exit keeps EXEC all-1s for WMMA.
// ---------------------------------------------------------------------------
__global__ void moe_gemm1(const _Float16* __restrict__ xh,
                          const _Float16* __restrict__ w13h,
                          const int* __restrict__ counts,
                          const int* __restrict__ idx,
                          _Float16* __restrict__ act) {
  const int e    = blockIdx.z;
  const int rows = counts[e];
  const int t0   = blockIdx.x * 32;
  if (t0 >= rows) return;

  const int wave = threadIdx.x >> 5;
  const int lane = threadIdx.x & 31;
  const int c0   = (blockIdx.y * 2 + wave) * 16;   // gate column tile in [0, I)
  const int half = lane >> 4;
  const int l15  = lane & 15;
  const int koff = half * 16;

  int s0 = t0 + l15, s1 = t0 + 16 + l15;
  int tok0 = idx[e * T_ + (s0 < rows - 1 ? s0 : rows - 1)];
  int tok1 = idx[e * T_ + (s1 < rows - 1 ? s1 : rows - 1)];
  const _Float16* x0 = xh + (size_t)tok0 * K_;
  const _Float16* x1 = xh + (size_t)tok1 * K_;
  const _Float16* wg = w13h + ((size_t)e * 2 * I_ + (c0 + l15)) * K_ + koff;
  const _Float16* wu = w13h + ((size_t)e * 2 * I_ + (I_ + c0 + l15)) * K_ + koff;

  v8f cg0 = {}, cg1 = {}, cu0 = {}, cu1 = {};
  for (int kb = 0; kb < K_; kb += 32) {
    v16h a0 = load_a_frag(x0, kb, half);
    v16h a1 = load_a_frag(x1, kb, half);
    v16h bg = *(const v16h*)(wg + kb);   // B frag: 16 contiguous K values
    v16h bu = *(const v16h*)(wu + kb);
    cg0 = WMMA_F16(a0, bg, cg0);
    cu0 = WMMA_F16(a0, bu, cu0);
    cg1 = WMMA_F16(a1, bg, cg1);
    cu1 = WMMA_F16(a1, bu, cu1);
  }
  // Fused SiLU(g)*u epilogue -> act[e, slot, i] (f16); fast rcp division
#pragma unroll
  for (int v = 0; v < 8; ++v) {
    int sa = t0 + v + 8 * half;          // C/D row: M = v + 8*(lane/16)
    int sb = sa + 16;
    float ga = cg0[v], ua = cu0[v];
    float gb = cg1[v], ub = cu1[v];
    float va = __fdividef(ga, 1.f + __expf(-ga)) * ua;
    float vb = __fdividef(gb, 1.f + __expf(-gb)) * ub;
    act[((size_t)e * T_ + sa) * I_ + (c0 + l15)] = (_Float16)va;
    act[((size_t)e * T_ + sb) * I_ + (c0 + l15)] = (_Float16)vb;
  }
}

// ---------------------------------------------------------------------------
// GEMM2 (gathered): y = act @ W2_e^T, scale by gate, scatter-add into out.
// Grid: (T/32, K/32, E), block = 64 (2 waves), 2 token tiles per wave.
// ---------------------------------------------------------------------------
__global__ void moe_gemm2(const _Float16* __restrict__ act,
                          const _Float16* __restrict__ w2h,
                          const int* __restrict__ counts,
                          const int* __restrict__ idx,
                          const float* __restrict__ gateg,
                          float* __restrict__ out) {
  const int e    = blockIdx.z;
  const int rows = counts[e];
  const int t0   = blockIdx.x * 32;
  if (t0 >= rows) return;

  const int wave = threadIdx.x >> 5;
  const int lane = threadIdx.x & 31;
  const int n0   = (blockIdx.y * 2 + wave) * 16;   // output column tile in [0, K)
  const int half = lane >> 4;
  const int l15  = lane & 15;
  const int koff = half * 16;

  const _Float16* a0 = act + ((size_t)e * T_ + (t0 + l15)) * I_;
  const _Float16* a1 = a0 + (size_t)16 * I_;
  const _Float16* w  = w2h + ((size_t)e * K_ + (n0 + l15)) * I_ + koff;

  v8f c0 = {}, c1 = {};
  for (int ib = 0; ib < I_; ib += 32) {
    v16h f0 = load_a_frag(a0, ib, half);
    v16h f1 = load_a_frag(a1, ib, half);
    v16h b  = *(const v16h*)(w + ib);
    c0 = WMMA_F16(f0, b, c0);
    c1 = WMMA_F16(f1, b, c1);
  }
  // Combine: out[token, n] += gate * y  (per-lane predication after WMMAs OK)
#pragma unroll
  for (int v = 0; v < 8; ++v) {
    int sa = t0 + v + 8 * half;
    int sb = sa + 16;
    if (sa < rows) {
      int   tok  = idx[e * T_ + sa];
      float gate = gateg[e * T_ + sa];
      atomicAdd(out + (size_t)tok * K_ + (n0 + l15), c0[v] * gate);
    }
    if (sb < rows) {
      int   tok  = idx[e * T_ + sb];
      float gate = gateg[e * T_ + sb];
      atomicAdd(out + (size_t)tok * K_ + (n0 + l15), c1[v] * gate);
    }
  }
}

// ---------------------------------------------------------------------------
extern "C" void kernel_launch(void* const* d_in, const int* in_sizes, int n_in,
                              void* d_out, int out_size, void* d_ws, size_t ws_size,
                              hipStream_t stream) {
  (void)in_sizes; (void)n_in; (void)out_size; (void)ws_size;
  const float* x       = (const float*)d_in[0];
  const float* logits  = (const float*)d_in[1];
  const int*   w13     = (const int*)d_in[2];
  const int*   w2      = (const int*)d_in[3];
  const float* alpha13 = (const float*)d_in[4];
  const float* alpha2  = (const float*)d_in[5];
  float*       out     = (float*)d_out;

  // Workspace layout (256B-aligned slices): ~141 MB total
  char* ws = (char*)d_ws;
  size_t off = 0;
  auto carve = [&](size_t bytes) {
    char* p = ws + off;
    off = (off + bytes + 255) & ~(size_t)255;
    return p;
  };
  int*      counts = (int*)carve((size_t)E_ * sizeof(int));
  int*      idx    = (int*)carve((size_t)E_ * T_ * sizeof(int));
  float*    gateg  = (float*)carve((size_t)E_ * T_ * sizeof(float));
  _Float16* w13h   = (_Float16*)carve((size_t)E_ * 2 * I_ * K_ * sizeof(_Float16));
  _Float16* w2h    = (_Float16*)carve((size_t)E_ * K_ * I_ * sizeof(_Float16));
  _Float16* xh     = (_Float16*)carve((size_t)T_ * K_ * sizeof(_Float16));
  _Float16* act    = (_Float16*)carve((size_t)E_ * T_ * I_ * sizeof(_Float16));

  const int nOut = T_ * K_;
  zero_kernel<<<(nOut + 255) / 256, 256, 0, stream>>>(out, nOut, counts);
  routing_kernel<<<(T_ + 127) / 128, 128, 0, stream>>>(logits, counts, idx, gateg);

  const long n13 = (long)E_ * 2 * I_ * K_;
  const long n2  = (long)E_ * K_ * I_;
  const long nx  = (long)T_ * K_;
  dequant13_kernel<<<(int)((n13 / 4 + 255) / 256), 256, 0, stream>>>(w13, alpha13, w13h);
  dequant2_kernel<<<(int)((n2 / 4 + 255) / 256), 256, 0, stream>>>(w2, alpha2, w2h);
  xcast_kernel<<<(int)((nx / 4 + 255) / 256), 256, 0, stream>>>(x, xh);

  moe_gemm1<<<dim3(T_ / 32, I_ / 32, E_), 64, 0, stream>>>(xh, w13h, counts, idx, act);
  moe_gemm2<<<dim3(T_ / 32, K_ / 32, E_), 64, 0, stream>>>(act, w2h, counts, idx, gateg, out);
}
